// SelfAttention_19851338842369
// MI455X (gfx1250) — matrix operand order repaired and checked by reading, compiled, not run on hardware
//
#include <hip/hip_runtime.h>
#include <math.h>

// ---------- types ----------
typedef __attribute__((ext_vector_type(16))) __bf16          v16bf;
typedef __attribute__((ext_vector_type(16))) unsigned short  v16us;
typedef __attribute__((ext_vector_type(8)))  float           v8f;

__device__ __forceinline__ unsigned short f32_to_bf16(float f) {
  unsigned int u = __float_as_uint(f);
  unsigned int r = u + 0x7fffu + ((u >> 16) & 1u);   // round-to-nearest-even
  return (unsigned short)(r >> 16);
}

// =====================================================================
// Tiled WMMA GEMM:  C[R x N] = scale * A[R x K] * op(B)
//   B_NT = true : op(B) = B^T with B stored [N x K] row-major (dot of rows)
//   B_NT = false: op(B) = B   with B stored [K x N] row-major (bf16 only)
// Block tile 128x128, BK=32, 256 threads = 8 wave32.
// Wave w: rows (w>>1)*32 .. +32, cols (w&1)*64 .. +64
//   -> 2 A-fragments, 4 B-fragments, 8 WMMAs per K-step per wave.
// =====================================================================
template <bool A_F32, bool B_F32, bool B_NT, bool OUT_BF16>
__global__ void __launch_bounds__(256)
wmma_gemm_kernel(const void* __restrict__ Ap, const void* __restrict__ Bp,
                 void* __restrict__ Cp, int K, int lda, int ldb, int ldc,
                 float scale, size_t bsA, size_t bsB, size_t bsC) {
  constexpr int BM = 128, BN = 128, BK = 32, PAD = 8;
  __shared__ unsigned short sA[BM][BK + PAD];
  __shared__ unsigned short sB[BN][BK + PAD];   // always [n][k]

  const int tid   = threadIdx.x;
  const int lane  = tid & 31;
  const int wave  = tid >> 5;
  const int z     = blockIdx.z;

  const int rowBase = blockIdx.y * BM;
  const int colBase = blockIdx.x * BN;

  const int wr    = wave >> 1;    // 0..3  (32-row slice)
  const int wc    = wave & 1;     // 0..1  (64-col slice)
  const int lrow  = lane & 15;
  const int khalf = lane >> 4;    // 0 or 1

  v8f acc[2][4] = {{{}, {}, {}, {}}, {{}, {}, {}, {}}};

  union U128 { uint4 v; unsigned short e[8]; };

  for (int k0 = 0; k0 < K; k0 += BK) {
    // ---------------- stage A tile: 128 rows x 32 k, contiguous K --------
    {
      const int r   = tid >> 1;          // 0..127
      const int c16 = (tid & 1) * 16;    // 0 or 16
      if constexpr (A_F32) {
        const float* a = (const float*)Ap + (size_t)z * bsA +
                         (size_t)(rowBase + r) * lda + k0 + c16;
        if (k0 + BK < K) __builtin_prefetch(a + BK, 0, 1);
        const float4 f0 = *(const float4*)(a + 0);
        const float4 f1 = *(const float4*)(a + 4);
        const float4 f2 = *(const float4*)(a + 8);
        const float4 f3 = *(const float4*)(a + 12);
        unsigned short* d = &sA[r][c16];
        d[0]  = f32_to_bf16(f0.x); d[1]  = f32_to_bf16(f0.y);
        d[2]  = f32_to_bf16(f0.z); d[3]  = f32_to_bf16(f0.w);
        d[4]  = f32_to_bf16(f1.x); d[5]  = f32_to_bf16(f1.y);
        d[6]  = f32_to_bf16(f1.z); d[7]  = f32_to_bf16(f1.w);
        d[8]  = f32_to_bf16(f2.x); d[9]  = f32_to_bf16(f2.y);
        d[10] = f32_to_bf16(f2.z); d[11] = f32_to_bf16(f2.w);
        d[12] = f32_to_bf16(f3.x); d[13] = f32_to_bf16(f3.y);
        d[14] = f32_to_bf16(f3.z); d[15] = f32_to_bf16(f3.w);
      } else {
        const unsigned short* a = (const unsigned short*)Ap + (size_t)z * bsA +
                                  (size_t)(rowBase + r) * lda + k0 + c16;
        if (k0 + BK < K) __builtin_prefetch(a + BK, 0, 1);
        *(uint4*)&sA[r][c16]     = *(const uint4*)(a);
        *(uint4*)&sA[r][c16 + 8] = *(const uint4*)(a + 8);
      }
    }
    // ---------------- stage B tile into sB[n][k] --------------------------
    if constexpr (B_NT) {
      const int n   = tid >> 1;
      const int c16 = (tid & 1) * 16;
      if constexpr (B_F32) {
        const float* b = (const float*)Bp + (size_t)z * bsB +
                         (size_t)(colBase + n) * ldb + k0 + c16;
        if (k0 + BK < K) __builtin_prefetch(b + BK, 0, 1);
        const float4 f0 = *(const float4*)(b + 0);
        const float4 f1 = *(const float4*)(b + 4);
        const float4 f2 = *(const float4*)(b + 8);
        const float4 f3 = *(const float4*)(b + 12);
        unsigned short* d = &sB[n][c16];
        d[0]  = f32_to_bf16(f0.x); d[1]  = f32_to_bf16(f0.y);
        d[2]  = f32_to_bf16(f0.z); d[3]  = f32_to_bf16(f0.w);
        d[4]  = f32_to_bf16(f1.x); d[5]  = f32_to_bf16(f1.y);
        d[6]  = f32_to_bf16(f1.z); d[7]  = f32_to_bf16(f1.w);
        d[8]  = f32_to_bf16(f2.x); d[9]  = f32_to_bf16(f2.y);
        d[10] = f32_to_bf16(f2.z); d[11] = f32_to_bf16(f2.w);
        d[12] = f32_to_bf16(f3.x); d[13] = f32_to_bf16(f3.y);
        d[14] = f32_to_bf16(f3.z); d[15] = f32_to_bf16(f3.w);
      } else {
        const unsigned short* b = (const unsigned short*)Bp + (size_t)z * bsB +
                                  (size_t)(colBase + n) * ldb + k0 + c16;
        if (k0 + BK < K) __builtin_prefetch(b + BK, 0, 1);
        *(uint4*)&sB[n][c16]     = *(const uint4*)(b);
        *(uint4*)&sB[n][c16 + 8] = *(const uint4*)(b + 8);
      }
    } else {
      // B stored [K x N] (bf16): load 16 contiguous cols, transpose into sB
      const int kk = tid >> 3;           // 0..31
      const int nn = (tid & 7) * 16;     // 0..112
      const unsigned short* b = (const unsigned short*)Bp + (size_t)z * bsB +
                                (size_t)(k0 + kk) * ldb + colBase + nn;
      if (k0 + BK < K) __builtin_prefetch(b + (size_t)BK * ldb, 0, 1);
      U128 u0, u1;
      u0.v = *(const uint4*)(b);
      u1.v = *(const uint4*)(b + 8);
      #pragma unroll
      for (int i = 0; i < 8; ++i) sB[nn + i][kk]     = u0.e[i];
      #pragma unroll
      for (int i = 0; i < 8; ++i) sB[nn + 8 + i][kk] = u1.e[i];
    }
    __syncthreads();

    // ---------------- build A fragments (16-bit A 16x32 layout) -----------
    v16bf afrag[2];
    #pragma unroll
    for (int i = 0; i < 2; ++i) {
      const int am = wr * 32 + i * 16 + lrow;
      v16us au;
      #pragma unroll
      for (int p = 0; p < 8; ++p) {
        const int kA = 2 * (p & 3) + 8 * khalf + 16 * (p >> 2);
        const unsigned int w = *(const unsigned int*)&sA[am][kA];
        au[2 * p]     = (unsigned short)w;
        au[2 * p + 1] = (unsigned short)(w >> 16);
      }
      afrag[i] = __builtin_bit_cast(v16bf, au);
    }

    // ---------------- B fragments + 8 WMMAs per K-step --------------------
    #pragma unroll
    for (int j = 0; j < 4; ++j) {
      const int bn = wc * 64 + j * 16 + lrow;
      v16us bu;
      #pragma unroll
      for (int p = 0; p < 8; ++p) {
        const int kB = 2 * (p & 3) + 8 * khalf + 16 * (p >> 2);
        const unsigned int w = *(const unsigned int*)&sB[bn][kB];
        bu[2 * p]     = (unsigned short)w;
        bu[2 * p + 1] = (unsigned short)(w >> 16);
      }
      const v16bf bfrag = __builtin_bit_cast(v16bf, bu);
      acc[0][j] = __builtin_amdgcn_wmma_f32_16x16x32_bf16(
          false, afrag[0], false, bfrag, (short)0, acc[0][j], false, false);
      acc[1][j] = __builtin_amdgcn_wmma_f32_16x16x32_bf16(
          false, afrag[1], false, bfrag, (short)0, acc[1][j], false, false);
    }
    __syncthreads();
  }

  // ---------------- write back (C/D layout: VGPR r -> row r + 8*khalf) ----
  #pragma unroll
  for (int i = 0; i < 2; ++i) {
    #pragma unroll
    for (int j = 0; j < 4; ++j) {
      const int cn = colBase + wc * 64 + j * 16 + lrow;
      #pragma unroll
      for (int r = 0; r < 8; ++r) {
        const int cm = rowBase + wr * 32 + i * 16 + r + 8 * khalf;
        const float val = acc[i][j][r] * scale;
        if constexpr (OUT_BF16)
          ((unsigned short*)Cp)[(size_t)z * bsC + (size_t)cm * ldc + cn] =
              f32_to_bf16(val);
        else
          ((float*)Cp)[(size_t)z * bsC + (size_t)cm * ldc + cn] = val;
      }
    }
  }
}

// =====================================================================
// Column softmax over the query axis (axis=1 of [B, Sq, Sk]).
// One thread per key column; row-wise accesses are coalesced across lanes.
// =====================================================================
__global__ void __launch_bounds__(256)
colsoftmax_kernel(const float* __restrict__ S, unsigned short* __restrict__ A,
                  int Sq, int Sk) {
  const int k = blockIdx.x * blockDim.x + threadIdx.x;
  const int b = blockIdx.y;
  const float* col = S + (size_t)b * Sq * Sk + k;
  float m = -3.402823466e38f, sum = 0.0f;
  for (int q = 0; q < Sq; ++q) {
    const float x  = col[(size_t)q * Sk];
    const float nm = fmaxf(m, x);
    sum = sum * __expf(m - nm) + __expf(x - nm);
    m = nm;
  }
  const float inv = 1.0f / sum;
  unsigned short* out = A + (size_t)b * Sq * Sk + k;
  for (int q = 0; q < Sq; ++q) {
    const float x = col[(size_t)q * Sk];
    out[(size_t)q * Sk] = f32_to_bf16(__expf(x - m) * inv);
  }
}

// =====================================================================
extern "C" void kernel_launch(void* const* d_in, const int* in_sizes, int n_in,
                              void* d_out, int out_size, void* d_ws,
                              size_t ws_size, hipStream_t stream) {
  (void)in_sizes; (void)n_in; (void)out_size; (void)ws_size;
  constexpr int B = 4, S = 2048, M = 2048, K = 2048, V = 2048;

  const float* q  = (const float*)d_in[0];
  const float* k  = (const float*)d_in[1];
  const float* v  = (const float*)d_in[2];
  const float* Wq = (const float*)d_in[3];
  const float* Wk = (const float*)d_in[4];
  const float* Wv = (const float*)d_in[5];

  // workspace layout: qp/kp/vp bf16 (3 x 32MiB), scores f32 (64MiB),
  // attn bf16 (32MiB) -> 192 MiB total
  const size_t nBSK = (size_t)B * S * K;
  unsigned short* qp     = (unsigned short*)d_ws;
  unsigned short* kp     = qp + nBSK;
  unsigned short* vp     = kp + nBSK;
  float*          scores = (float*)(vp + nBSK);
  unsigned short* attn   = (unsigned short*)(scores + (size_t)B * S * S);

  const dim3 blk(256);

  // ---- 1) projections: [B*S x M] * W[N x M]^T  -> bf16 ------------------
  {
    const dim3 g(K / 128, (B * S) / 128, 1);
    wmma_gemm_kernel<true, true, true, true><<<g, blk, 0, stream>>>(
        q, Wq, qp, M, M, M, K, 1.0f, 0, 0, 0);
    wmma_gemm_kernel<true, true, true, true><<<g, blk, 0, stream>>>(
        k, Wk, kp, M, M, M, K, 1.0f, 0, 0, 0);
    wmma_gemm_kernel<true, true, true, true><<<g, blk, 0, stream>>>(
        v, Wv, vp, M, M, M, V, 1.0f, 0, 0, 0);
  }

  // ---- 2) scores = qp . kp^T * K^-0.5  (batched, f32 out) ---------------
  {
    const dim3 g(S / 128, S / 128, B);
    const float scale = 1.0f / sqrtf((float)K);
    wmma_gemm_kernel<false, false, true, false><<<g, blk, 0, stream>>>(
        qp, kp, scores, K, K, K, S, scale,
        (size_t)S * K, (size_t)S * K, (size_t)S * S);
  }

  // ---- 3) softmax over the query axis (axis=1), f32 -> bf16 -------------
  colsoftmax_kernel<<<dim3(S / 256, B), blk, 0, stream>>>(scores, attn, S, S);

  // ---- 4) out = attn [Sq x Sk] . vp [Sk x V]  (NN, f32 out) -------------
  {
    const dim3 g(V / 128, S / 128, B);
    wmma_gemm_kernel<false, false, false, false><<<g, blk, 0, stream>>>(
        attn, vp, (float*)d_out, S, S, V, V, 1.0f,
        (size_t)S * S, (size_t)S * V, (size_t)S * V);
  }
}
